// MaskedSelfAttention_89120571392401
// MI455X (gfx1250) — matrix operand order repaired and checked by reading, not compiled, not run on hardware
//
#include <hip/hip_runtime.h>
#include <hip/hip_bf16.h>

// ---------------------------------------------------------------------------
// MI455X (gfx1250) masked self-attention: QKV GEMM + RoPE + causal flash
// attention + output GEMM. All matmuls via v_wmma_f32_16x16x32_bf16; GEMM
// B-tiles staged in LDS with gfx1250 async global->LDS loads (ASYNCcnt),
// double-buffered.
// ---------------------------------------------------------------------------

#define T_SEQ   2048
#define N_B     2
#define N_H     16
#define HDIM    64
#define DMODEL  1024

typedef __attribute__((ext_vector_type(16))) __bf16 bf16x16;
typedef __attribute__((ext_vector_type(8)))  __bf16 bf16x8;
typedef __attribute__((ext_vector_type(8)))  float  f32x8;

// Load 16 bf16 as two 16B-aligned halves (works for global and LDS pointers).
__device__ __forceinline__ bf16x16 load16(const __bf16* p0, const __bf16* p1) {
    bf16x8 a = *reinterpret_cast<const bf16x8*>(p0);
    bf16x8 b = *reinterpret_cast<const bf16x8*>(p1);
    bf16x16 r;
#pragma unroll
    for (int i = 0; i < 8; ++i) { r[i] = a[i]; r[i + 8] = b[i]; }
    return r;
}

__device__ __forceinline__ f32x8 wmma_bf16(bf16x16 a, bf16x16 b, f32x8 c) {
    return __builtin_amdgcn_wmma_f32_16x16x32_bf16(
        /*neg_a=*/false, a, /*neg_b=*/false, b,
        /*c_mod=*/(short)0, c, /*reuse_a=*/false, /*reuse_b=*/false);
}

// gfx1250 async global->LDS copy (16B per lane), tracked by ASYNCcnt.
// lds_addr = low 32 bits of the generic LDS pointer (wave-relative offset).
__device__ __forceinline__ void async_load_b128(uint32_t lds_addr,
                                                const void* gaddr) {
    asm volatile("global_load_async_to_lds_b128 %0, %1, off"
                 :: "v"(lds_addr), "v"(gaddr)
                 : "memory");
}

// ---------------------------------------------------------------------------
// Prep kernels
// ---------------------------------------------------------------------------
__global__ void cvt_f32_to_bf16_kernel(const float* __restrict__ in,
                                       __bf16* __restrict__ out, int n) {
    int i = blockIdx.x * blockDim.x + threadIdx.x;
    if (i < n) out[i] = (__bf16)in[i];
}

// W[K,N] f32 (row-major)  ->  Wt[N,K] bf16 (row-major); i indexes output.
__global__ void transpose_w_kernel(const float* __restrict__ W,
                                   __bf16* __restrict__ Wt, int K, int N) {
    int i = blockIdx.x * blockDim.x + threadIdx.x;
    if (i >= K * N) return;
    int k = i % K;
    int n = i / K;
    Wt[(size_t)n * K + k] = (__bf16)W[(size_t)k * N + n];
}

// ---------------------------------------------------------------------------
// bf16 WMMA GEMM:  C[M,N] f32 = A[M,K] bf16 (row-major) * Bt[N,K] bf16
// Block = 4 waves, 128x64 C tile. Each wave owns 32x64 (8 WMMA tiles).
// The 64x32 B tile is shared by all waves: staged in LDS via async
// global->LDS b128 copies, double-buffered across K chunks.
// ---------------------------------------------------------------------------
#define BROW_PAD 40   // row stride in elements (80B: 16B-aligned, bank-spread)

__global__ __launch_bounds__(128) void gemm_bf16_kernel(
    const __bf16* __restrict__ A, const __bf16* __restrict__ Bt,
    float* __restrict__ C, int M, int N, int K) {
    __shared__ __align__(16) __bf16 lds_b[2][64 * BROW_PAD];

    const int tid  = threadIdx.x;
    const int lane = tid & 31;
    const int wave = tid >> 5;
    const int Ml   = lane & 15;
    const int half = lane >> 4;
    const int m0   = blockIdx.x * 128 + wave * 32;
    const int n0   = blockIdx.y * 64;

    // Producer: thread tid fills B row n=tid/2, 16 elements at col (tid&1)*16.
    const int prow = tid >> 1;
    const int pcol = (tid & 1) * 16;
    const __bf16* gb0 = Bt + (size_t)(n0 + prow) * K + pcol;
    const uint32_t lds0 =
        (uint32_t)(uintptr_t)(&lds_b[0][0]) + (uint32_t)(prow * BROW_PAD + pcol) * 2;
    const uint32_t buf_stride = (uint32_t)(64 * BROW_PAD * 2);

    const __bf16* arow0 = A + (size_t)(m0 + Ml) * K;
    const __bf16* arow1 = A + (size_t)(m0 + 16 + Ml) * K;

    f32x8 acc[8] = {};
    const int nchunks = K / 32;

    // Prefetch chunk 0 into buffer 0 (two 16B async copies per thread).
    async_load_b128(lds0, gb0);
    async_load_b128(lds0 + 16, (const char*)gb0 + 16);

    for (int kc = 0; kc < nchunks; ++kc) {
        const int buf = kc & 1;
        if (kc + 1 < nchunks) {
            const char* g  = (const char*)gb0 + (size_t)(kc + 1) * 64;
            uint32_t    la = lds0 + (uint32_t)(buf ^ 1) * buf_stride;
            async_load_b128(la, g);
            async_load_b128(la + 16, g + 16);
            // In-order completion: <=2 outstanding => current chunk landed.
            asm volatile("s_wait_asynccnt 0x2" ::: "memory");
        } else {
            asm volatile("s_wait_asynccnt 0x0" ::: "memory");
        }
        __syncthreads();

        const int k = kc * 32;
        // A operands: M = lane%16, K(e) = (e<8?0:16) + 8*half + e%8
        bf16x16 a0 = load16(arow0 + k + half * 8, arow0 + k + 16 + half * 8);
        bf16x16 a1 = load16(arow1 + k + half * 8, arow1 + k + 16 + half * 8);
#pragma unroll
        for (int nt = 0; nt < 4; ++nt) {
            // B operand from LDS: N = nt*16+lane%16, K(e) = (e<8?0:16) + 8*half + e%8
            const __bf16* bp =
                &lds_b[buf][(nt * 16 + Ml) * BROW_PAD + half * 8];
            bf16x16 b = load16(bp, bp + 16);
            acc[nt]     = wmma_bf16(a0, b, acc[nt]);
            acc[4 + nt] = wmma_bf16(a1, b, acc[4 + nt]);
        }
    }

#pragma unroll
    for (int mt = 0; mt < 2; ++mt)
#pragma unroll
        for (int nt = 0; nt < 4; ++nt)
#pragma unroll
            for (int r = 0; r < 8; ++r)
                C[(size_t)(m0 + mt * 16 + r + 8 * half) * N + n0 + nt * 16 + Ml] =
                    acc[mt * 4 + nt][r];
}

// ---------------------------------------------------------------------------
// RoPE + repack: qkv f32 [B*T, 3*DMODEL] ->
//   Qr, Kr bf16 [BH, T, 64] row-major ; Vt bf16 [BH, 64, T] (transposed)
// One thread per (b,h,t,j), j in 0..31 handles the rotation pair (j, j+32).
// ---------------------------------------------------------------------------
__global__ void rope_pack_kernel(const float* __restrict__ qkv,
                                 __bf16* __restrict__ Qr,
                                 __bf16* __restrict__ Kr,
                                 __bf16* __restrict__ Vt) {
    int i = blockIdx.x * blockDim.x + threadIdx.x;  // B*T*H*32 threads
    int j = i & 31;
    int h = (i >> 5) & (N_H - 1);
    int t = (i >> 9) & (T_SEQ - 1);
    int b = i >> 20;

    const float* row = qkv + (size_t)(b * T_SEQ + t) * (3 * DMODEL);
    int col = h * HDIM + j;
    float q1 = row[col],            q2 = row[col + 32];
    float k1 = row[DMODEL + col],   k2 = row[DMODEL + col + 32];
    float v1 = row[2*DMODEL + col], v2 = row[2*DMODEL + col + 32];

    // inv_freq = 10000^(-2j/64) = exp2(-j * log2(10000)/32)
    const float kF = 13.287712379549449f / 32.0f;
    float f = exp2f(-(float)j * kF);
    float s, c;
    sincosf((float)t * f, &s, &c);

    float qr1 = q1 * c - q2 * s, qr2 = q2 * c + q1 * s;
    float kr1 = k1 * c - k2 * s, kr2 = k2 * c + k1 * s;

    int bh = b * N_H + h;
    size_t qk = ((size_t)bh * T_SEQ + t) * HDIM + j;
    Qr[qk]      = (__bf16)qr1;  Qr[qk + 32] = (__bf16)qr2;
    Kr[qk]      = (__bf16)kr1;  Kr[qk + 32] = (__bf16)kr2;
    size_t vb = (size_t)bh * HDIM * T_SEQ + (size_t)t;
    Vt[vb + (size_t)j * T_SEQ]        = (__bf16)v1;
    Vt[vb + (size_t)(j + 32) * T_SEQ] = (__bf16)v2;
}

// ---------------------------------------------------------------------------
// Causal flash attention. One wave = 16 query rows; streams keys in 32-blocks.
// S tiles via WMMA(Q,K^T); running (m,l) kept replicated-per-half so the
// C-layout rescale needs no shuffles; P transposed C->A layout via LDS.
// attn output written bf16 [B, T, DMODEL] for the final GEMM.
// ---------------------------------------------------------------------------
__global__ __launch_bounds__(128) void flash_attn_kernel(
    const __bf16* __restrict__ Qr, const __bf16* __restrict__ Kr,
    const __bf16* __restrict__ Vt, __bf16* __restrict__ attn) {
    __shared__ __align__(16) __bf16 lds_p[4][16 * 32];

    const int lane = threadIdx.x & 31;
    const int wave = threadIdx.x >> 5;
    const int Ml   = lane & 15;
    const int half = lane >> 4;
    const int bh   = blockIdx.x >> 5;   // 32 query-groups per (b,h)
    const int grp  = blockIdx.x & 31;
    const int q0   = (grp * 4 + wave) * 16;
    const int b    = bh >> 4;
    const int h    = bh & 15;

    const float scale = 0.125f;                 // 1/sqrt(64)
    const float LOG2E = 1.4426950408889634f;

    // Q A-operands for head-dim chunks [0,32) and [32,64)
    const __bf16* Qp = Qr + ((size_t)bh * T_SEQ + q0 + Ml) * HDIM;
    bf16x16 aq0 = load16(Qp + half * 8,      Qp + 16 + half * 8);
    bf16x16 aq1 = load16(Qp + 32 + half * 8, Qp + 48 + half * 8);

    f32x8 o[4] = {};
    float m8[8], l8[8];
#pragma unroll
    for (int r = 0; r < 8; ++r) { m8[r] = -1e30f; l8[r] = 0.0f; }

    __bf16* pl = lds_p[wave];

    for (int kb = 0; kb < q0 + 16; kb += 32) {
        // ---- S = (Q K^T) * scale, two 16-wide key tiles ----
        f32x8 s[2];
#pragma unroll
        for (int kt = 0; kt < 2; ++kt) {
            const __bf16* Kp =
                Kr + ((size_t)bh * T_SEQ + kb + kt * 16 + Ml) * HDIM;
            bf16x16 b0 = load16(Kp + half * 8,      Kp + 16 + half * 8);
            bf16x16 b1 = load16(Kp + 32 + half * 8, Kp + 48 + half * 8);
            f32x8 acc = {};
            acc = wmma_bf16(aq0, b0, acc);
            acc = wmma_bf16(aq1, b1, acc);
            s[kt] = acc;
        }
        // scale + causal mask (key > row -> -inf)
#pragma unroll
        for (int kt = 0; kt < 2; ++kt)
#pragma unroll
            for (int r = 0; r < 8; ++r) {
                float v  = s[kt][r] * scale;
                int  key = kb + kt * 16 + Ml;
                int  row = q0 + r + 8 * half;
                s[kt][r] = (key > row) ? -1e30f : v;
            }
        // ---- running max update (replicated within each 16-lane half) ----
        float alpha8[8];
#pragma unroll
        for (int r = 0; r < 8; ++r) {
            float t = fmaxf(s[0][r], s[1][r]);
            t = fmaxf(t, __shfl_xor(t, 1));
            t = fmaxf(t, __shfl_xor(t, 2));
            t = fmaxf(t, __shfl_xor(t, 4));
            t = fmaxf(t, __shfl_xor(t, 8));
            float mnew = fmaxf(m8[r], t);
            alpha8[r]  = exp2f((m8[r] - mnew) * LOG2E);
            m8[r]      = mnew;
        }
        // ---- P = exp(S - m); row sums; stage P (bf16) to LDS in [16][32] ----
#pragma unroll
        for (int r = 0; r < 8; ++r) {
            float p0 = exp2f((s[0][r] - m8[r]) * LOG2E);
            float p1 = exp2f((s[1][r] - m8[r]) * LOG2E);
            float rs = p0 + p1;
            rs += __shfl_xor(rs, 1);
            rs += __shfl_xor(rs, 2);
            rs += __shfl_xor(rs, 4);
            rs += __shfl_xor(rs, 8);
            l8[r] = l8[r] * alpha8[r] + rs;
            int row = r + 8 * half;
            pl[row * 32 + Ml]      = (__bf16)p0;
            pl[row * 32 + 16 + Ml] = (__bf16)p1;
        }
        // ---- rescale O tiles (row index matches r+8*half directly) ----
#pragma unroll
        for (int nt = 0; nt < 4; ++nt)
#pragma unroll
            for (int r = 0; r < 8; ++r) o[nt][r] *= alpha8[r];
        // ---- O += P @ V_block : A from LDS (A-layout), B from Vt ----
        bf16x16 ap = load16(pl + Ml * 32 + half * 8,
                            pl + Ml * 32 + 16 + half * 8);
#pragma unroll
        for (int nt = 0; nt < 4; ++nt) {
            const __bf16* Vp = Vt + ((size_t)bh * HDIM + nt * 16 + Ml) * T_SEQ
                                  + kb + half * 8;
            bf16x16 bv = load16(Vp, Vp + 16);
            o[nt] = wmma_bf16(ap, bv, o[nt]);
        }
    }

    // ---- normalize and write attn [B, T, DMODEL] as bf16 ----
    float inv8[8];
#pragma unroll
    for (int r = 0; r < 8; ++r) inv8[r] = 1.0f / l8[r];
#pragma unroll
    for (int nt = 0; nt < 4; ++nt)
#pragma unroll
        for (int r = 0; r < 8; ++r) {
            int row = q0 + r + 8 * half;
            attn[((size_t)b * T_SEQ + row) * DMODEL + h * HDIM + nt * 16 + Ml] =
                (__bf16)(o[nt][r] * inv8[r]);
        }
}

// ---------------------------------------------------------------------------
// Launch
// ---------------------------------------------------------------------------
extern "C" void kernel_launch(void* const* d_in, const int* in_sizes, int n_in,
                              void* d_out, int out_size, void* d_ws,
                              size_t ws_size, hipStream_t stream) {
    const float* x    = (const float*)d_in[0];   // [B,T,DMODEL]
    const float* Wqkv = (const float*)d_in[1];   // [DMODEL, 3*DMODEL]
    const float* Wout = (const float*)d_in[2];   // [DMODEL, DMODEL]
    float* out = (float*)d_out;                  // [B,T,DMODEL] f32

    const int BT = N_B * T_SEQ;                  // 4096 rows
    char* ws = (char*)d_ws;
    size_t off = 0;
    auto alloc = [&](size_t bytes) {
        void* p = ws + off;
        off += (bytes + 255) & ~(size_t)255;
        return p;
    };
    __bf16* xb     = (__bf16*)alloc((size_t)BT * DMODEL * 2);
    __bf16* wqkv_t = (__bf16*)alloc((size_t)3 * DMODEL * DMODEL * 2);
    __bf16* wout_t = (__bf16*)alloc((size_t)DMODEL * DMODEL * 2);
    float*  qkv    = (float*) alloc((size_t)BT * 3 * DMODEL * 4);
    __bf16* Qr     = (__bf16*)alloc((size_t)N_B * N_H * T_SEQ * HDIM * 2);
    __bf16* Kr     = (__bf16*)alloc((size_t)N_B * N_H * T_SEQ * HDIM * 2);
    __bf16* Vt     = (__bf16*)alloc((size_t)N_B * N_H * T_SEQ * HDIM * 2);
    __bf16* attn   = (__bf16*)alloc((size_t)BT * DMODEL * 2);

    // 1) stage operands as bf16 (weights transposed to [N,K])
    {
        int n = BT * DMODEL;
        cvt_f32_to_bf16_kernel<<<(n + 255) / 256, 256, 0, stream>>>(x, xb, n);
    }
    {
        int n = DMODEL * 3 * DMODEL;
        transpose_w_kernel<<<(n + 255) / 256, 256, 0, stream>>>(
            Wqkv, wqkv_t, DMODEL, 3 * DMODEL);
    }
    {
        int n = DMODEL * DMODEL;
        transpose_w_kernel<<<(n + 255) / 256, 256, 0, stream>>>(
            Wout, wout_t, DMODEL, DMODEL);
    }
    // 2) qkv = x @ Wqkv  (WMMA, async-LDS B staging)
    gemm_bf16_kernel<<<dim3(BT / 128, (3 * DMODEL) / 64), 128, 0, stream>>>(
        xb, wqkv_t, qkv, BT, 3 * DMODEL, DMODEL);
    // 3) RoPE + head-major repack (V transposed)
    {
        int n = N_B * T_SEQ * N_H * 32;
        rope_pack_kernel<<<n / 256, 256, 0, stream>>>(qkv, Qr, Kr, Vt);
    }
    // 4) causal flash attention (WMMA)
    flash_attn_kernel<<<N_B * N_H * 32, 128, 0, stream>>>(Qr, Kr, Vt, attn);
    // 5) out = attn @ Wout  (WMMA, async-LDS B staging)
    gemm_bf16_kernel<<<dim3(BT / 128, DMODEL / 64), 128, 0, stream>>>(
        attn, wout_t, out, BT, DMODEL, DMODEL);
}
